// NewAttentionBlock_7069516169836
// MI455X (gfx1250) — matrix operand order repaired and checked by reading, not compiled, not run on hardware
//
#include <hip/hip_runtime.h>
#include <hip/hip_bf16.h>
#include <stdint.h>

typedef __attribute__((ext_vector_type(16))) __bf16 v16bf;
typedef __attribute__((ext_vector_type(8)))  float  v8f;
typedef __attribute__((ext_vector_type(4)))  float  v4f;
typedef __attribute__((ext_vector_type(2)))  float  v2f;

#define B_     8
#define H_     16
#define QL     16
#define SP     4096
#define D_     256
#define SCALE  0.0625f
#define NEGBIG (-3.0e38f)
#define CK     128   // keys per chunk per workgroup (8 waves x 16)
#define NTHREADS 256

// Compute one 16x16 score tile: C += A(16x32 bf16 slices of Q) x B(32x16 bf16 slices of K^T)
// over the full D=256, via 8 x v_wmma_f32_16x16x32_bf16.
// B-frag (32x16, bf16): lane holds column n=lane&15; lanes 0-15 hold K=0..7,16..23,
// lanes 16-31 hold K=8..15,24..31 (same order as the Q A-frag).
__device__ __forceinline__ v8f qk_tile(const v16bf qa[8], const float* __restrict__ krow, int hi) {
  v8f sacc = {0.f,0.f,0.f,0.f,0.f,0.f,0.f,0.f};
  #pragma unroll
  for (int s = 0; s < 8; ++s) {
    const float* kp = krow + 32*s + hi*8;
    v4f f0 = *(const v4f*)(kp);
    v4f f1 = *(const v4f*)(kp + 4);
    v4f f2 = *(const v4f*)(kp + 16);
    v4f f3 = *(const v4f*)(kp + 20);
    v16bf bk;
    #pragma unroll
    for (int j = 0; j < 4; ++j) {
      bk[j]      = (__bf16)f0[j];
      bk[4 + j]  = (__bf16)f1[j];
      bk[8 + j]  = (__bf16)f2[j];
      bk[12 + j] = (__bf16)f3[j];
    }
    sacc = __builtin_amdgcn_wmma_f32_16x16x32_bf16(false, qa[s], false, bk,
                                                   (short)0, sacc, false, false);
  }
  return sacc;
}

__global__ __launch_bounds__(NTHREADS) void fa_partial(
    const float* __restrict__ Q,  const float* __restrict__ Kp,
    const float* __restrict__ Vp, const float* __restrict__ Ka,
    const float* __restrict__ Va, const unsigned char* __restrict__ mask,
    float* __restrict__ out, float* __restrict__ Opart, float* __restrict__ mlpart,
    int NS)
{
  __shared__ float sc[QL][CK + 4];       // scores -> probabilities
  __shared__ float red[QL][17];          // per-row segment reduction scratch
  __shared__ float mrow[QL], lrow[QL], arow[QL];

  const int tid  = threadIdx.x;
  const int wave = tid >> 5;
  const int lane = tid & 31;
  const int hi   = lane >> 4;            // lane half (K-split for frags)
  const int n    = lane & 15;            // column / row index inside a tile
  const int bh   = blockIdx.x / NS;
  const int sp   = blockIdx.x % NS;
  const int b    = bh / H_;
  const int rrow = tid >> 4;             // softmax: row handled by this thread
  const int rseg = tid & 15;             // softmax: 8-wide segment within row

  // ---- Q as bf16 A-fragments (16x32 per slice), held in registers (64 VGPRs) ----
  v16bf qa[8];
  {
    const float* qrow = Q + ((size_t)bh * QL + n) * D_;
    #pragma unroll
    for (int s = 0; s < 8; ++s) {
      const float* p0 = qrow + 32*s + hi*8;
      const float* p1 = qrow + 32*s + 16 + hi*8;
      v4f f0 = *(const v4f*)(p0);
      v4f f1 = *(const v4f*)(p0 + 4);
      v4f f2 = *(const v4f*)(p1);
      v4f f3 = *(const v4f*)(p1 + 4);
      #pragma unroll
      for (int j = 0; j < 4; ++j) {
        qa[s][j]      = (__bf16)f0[j];
        qa[s][4 + j]  = (__bf16)f1[j];
        qa[s][8 + j]  = (__bf16)f2[j];
        qa[s][12 + j] = (__bf16)f3[j];
      }
    }
  }

  if (tid < QL) { mrow[tid] = NEGBIG; lrow[tid] = 0.f; }
  v8f o0 = {0.f,0.f,0.f,0.f,0.f,0.f,0.f,0.f};
  v8f o1 = {0.f,0.f,0.f,0.f,0.f,0.f,0.f,0.f};
  __syncthreads();

  const int keys_per_split = SP / NS;
  const int nchunks        = keys_per_split / CK;
  const int kbase          = sp * keys_per_split;
  const int niter          = nchunks + ((sp == 0) ? 1 : 0); // +1 = active-key chunk
  const int dimbase        = wave * 32;

  for (int c = 0; c < niter; ++c) {
    const bool active = (c == nchunks);

    // ---------------- QK phase: each wave -> one 16x16 score tile ----------------
    if (!active) {
      const int key = kbase + c * CK + wave * 16 + n;
      const float* krow = Kp + ((size_t)bh * SP + key) * D_;
      if (c + 1 < nchunks) __builtin_prefetch(krow + (size_t)CK * D_, 0, 0);
      v8f sacc = qk_tile(qa, krow, hi);
      const unsigned char* mb = mask + (size_t)b * QL * SP;
      #pragma unroll
      for (int v = 0; v < 8; ++v) {
        const int m = v + 8 * hi;          // C layout: M = vgpr + 8*(lane>=16)
        float scv = sacc[v] * SCALE;
        scv = mb[(size_t)m * SP + key] ? scv : NEGBIG;
        sc[m][wave * 16 + n] = scv;
      }
    } else {
      if (wave == 0) {
        const float* krow = Ka + ((size_t)bh * QL + n) * D_;
        v8f sacc = qk_tile(qa, krow, hi);
        #pragma unroll
        for (int v = 0; v < 8; ++v) sc[v + 8*hi][n] = sacc[v] * SCALE;
      } else {
        #pragma unroll
        for (int v = 0; v < 8; ++v) sc[v + 8*hi][wave * 16 + n] = NEGBIG;
      }
    }
    __syncthreads();

    // ---------------- online softmax over this chunk (parallel, all 256 threads) ----
    {
      float mx = NEGBIG;
      #pragma unroll
      for (int j = 0; j < 8; ++j) mx = fmaxf(mx, sc[rrow][rseg * 8 + j]);
      red[rrow][rseg] = mx;
    }
    __syncthreads();
    if (tid < QL) {
      float mx = NEGBIG;
      #pragma unroll
      for (int s2 = 0; s2 < 16; ++s2) mx = fmaxf(mx, red[tid][s2]);
      const float mold = mrow[tid];
      const float mnew = fmaxf(mold, mx);
      arow[tid] = __expf(mold - mnew);
      mrow[tid] = mnew;
    }
    __syncthreads();
    {
      const float mnew = mrow[rrow];
      float s = 0.f;
      #pragma unroll
      for (int j = 0; j < 8; ++j) {
        float p = __expf(sc[rrow][rseg * 8 + j] - mnew);
        sc[rrow][rseg * 8 + j] = p;
        s += p;
      }
      red[rrow][rseg] = s;
    }
    __syncthreads();
    if (tid < QL) {     // l-update overlaps with PV below (only needs red[] and arow)
      float s = 0.f;
      #pragma unroll
      for (int s2 = 0; s2 < 16; ++s2) s += red[tid][s2];
      lrow[tid] = lrow[tid] * arow[tid] + s;
    }

    // ---------------- rescale O, then PV: wave owns dims [32w, 32w+32) ----------------
    #pragma unroll
    for (int v = 0; v < 8; ++v) {
      const float al = arow[v + 8*hi];
      o0[v] *= al;
      o1[v] *= al;
    }
    // P x V with f32 WMMA (16x16x4): A = probs from LDS, B = V rows straight
    // from global memory (lanes 0-15 = 64B contiguous per row: coalesced).
    auto pv_step = [&](const float* __restrict__ Vb, int kb4) {
      const int r = kb4 * 4 + hi * 2;      // f32 frag K-split: lanes 0-15 K={0,1}, 16-31 K={2,3}
      v2f a;
      a[0] = sc[n][r];
      a[1] = sc[n][r + 1];
      const float* vp0 = Vb + (size_t)r * D_ + dimbase + n;
      v2f b0; b0[0] = vp0[0]; b0[1] = vp0[D_];
      o0 = __builtin_amdgcn_wmma_f32_16x16x4_f32(false, a, false, b0,
                                                 (short)0, o0, false, false);
      const float* vp1 = vp0 + 16;
      v2f b1; b1[0] = vp1[0]; b1[1] = vp1[D_];
      o1 = __builtin_amdgcn_wmma_f32_16x16x4_f32(false, a, false, b1,
                                                 (short)0, o1, false, false);
    };
    if (!active) {
      const float* Vb = Vp + ((size_t)bh * SP + (size_t)(kbase + c * CK)) * D_;
      if (c + 1 < nchunks)
        __builtin_prefetch(Vb + (size_t)CK * D_ + (size_t)lane * 16, 0, 0);
      #pragma unroll 4
      for (int kb4 = 0; kb4 < CK / 4; ++kb4) pv_step(Vb, kb4);
    } else {
      const float* Vb = Va + (size_t)bh * QL * D_;
      #pragma unroll
      for (int kb4 = 0; kb4 < 4; ++kb4) pv_step(Vb, kb4);
    }
    __syncthreads();   // sc / red reused next chunk
  }

  // ---------------- epilogue ----------------
  if (NS == 1) {
    float* orow = out + (size_t)bh * QL * D_;
    #pragma unroll
    for (int v = 0; v < 8; ++v) {
      const int m = v + 8*hi;
      const float inv = 1.0f / lrow[m];
      orow[(size_t)m * D_ + wave * 32 + n]      = o0[v] * inv;
      orow[(size_t)m * D_ + wave * 32 + 16 + n] = o1[v] * inv;
    }
  } else {
    float* op = Opart + (size_t)blockIdx.x * QL * D_;
    #pragma unroll
    for (int v = 0; v < 8; ++v) {
      const int m = v + 8*hi;
      op[(size_t)m * D_ + wave * 32 + n]      = o0[v];
      op[(size_t)m * D_ + wave * 32 + 16 + n] = o1[v];
    }
    if (tid < QL) {
      mlpart[(size_t)blockIdx.x * 32 + tid]      = mrow[tid];
      mlpart[(size_t)blockIdx.x * 32 + 16 + tid] = lrow[tid];
    }
  }
}

// Combine NS split partials per head: O = sum_s O_s * exp(m_s - M) / (sum_s l_s * exp(m_s - M))
__global__ __launch_bounds__(NTHREADS) void fa_merge(
    const float* __restrict__ Opart, const float* __restrict__ mlpart,
    float* __restrict__ out, int NS)
{
  const int bh = blockIdx.x;
  const int d  = threadIdx.x;              // 0..255
  for (int q = 0; q < QL; ++q) {
    float M = NEGBIG;
    for (int s = 0; s < NS; ++s)
      M = fmaxf(M, mlpart[(size_t)(bh * NS + s) * 32 + q]);
    float den = 0.f, acc = 0.f;
    for (int s = 0; s < NS; ++s) {
      const float w = __expf(mlpart[(size_t)(bh * NS + s) * 32 + q] - M);
      den += mlpart[(size_t)(bh * NS + s) * 32 + 16 + q] * w;
      acc += Opart[((size_t)(bh * NS + s) * QL + q) * D_ + d] * w;
    }
    out[((size_t)bh * QL + q) * D_ + d] = acc / den;
  }
}

extern "C" void kernel_launch(void* const* d_in, const int* in_sizes, int n_in,
                              void* d_out, int out_size, void* d_ws, size_t ws_size,
                              hipStream_t stream) {
  (void)in_sizes; (void)n_in; (void)out_size;
  const float* Q  = (const float*)d_in[0];
  const float* Kp = (const float*)d_in[1];
  const float* Vp = (const float*)d_in[2];
  const float* Ka = (const float*)d_in[3];
  const float* Va = (const float*)d_in[4];
  const unsigned char* mask = (const unsigned char*)d_in[5];
  float* out = (float*)d_out;

  const int BH = B_ * H_;
  auto need = [&](int ns) -> size_t {
    return (size_t)BH * ns * (QL * D_ + 32) * sizeof(float);
  };
  int NS = 1;
  if (ws_size >= need(4))      NS = 4;
  else if (ws_size >= need(2)) NS = 2;

  float* Opart  = (float*)d_ws;
  float* mlpart = Opart + (size_t)BH * NS * QL * D_;

  fa_partial<<<dim3(BH * NS), dim3(NTHREADS), 0, stream>>>(
      Q, Kp, Vp, Ka, Va, mask, out, Opart, mlpart, NS);
  if (NS > 1) {
    fa_merge<<<dim3(BH), dim3(NTHREADS), 0, stream>>>(Opart, mlpart, out, NS);
  }
}
